// GNNLayer_29111288332892
// MI455X (gfx1250) — matrix operand order repaired and checked by reading, not compiled, not run on hardware
//
#include <hip/hip_runtime.h>
#include <math.h>

// ---------------------------------------------------------------------------
// GAT layer for MI455X (gfx1250, wave32).
//   B=2, N=10000, D=128, H=4, HD=32, E=640000
// Phase 1: projection GEMM via v_wmma_f32_16x16x32_bf16 (fp32 accum);
//          W operand staged to LDS via GLOBAL_LOAD_ASYNC_TO_LDS_B128.
// Phase 2: per-node logit scalars (factor edge dot-products out of edge loop).
// Phase 3: global per-(b,h) max (ordered-uint atomicMax) + global_prefetch.
// Phase 4: exp + softmax denominator scatter (global_atomic_add_f32, L2-resident).
// Phase 5: message scatter-add (1 wave per (b,edge), 128 f32 atomics).
// Phase 6: residual is pre-seeded; LayerNorm + ReLU in place on d_out.
// ---------------------------------------------------------------------------

typedef __attribute__((ext_vector_type(16))) __bf16 v16bf;
typedef __attribute__((ext_vector_type(8)))  float  v8f;
typedef __attribute__((ext_vector_type(4)))  int    v4i_t;

#define DD  128
#define HH  4
#define HDD 32

#if defined(__HIP_DEVICE_COMPILE__) &&                                   \
    __has_builtin(__builtin_amdgcn_global_load_async_to_lds_b128) &&     \
    __has_builtin(__builtin_amdgcn_s_wait_asynccnt)
#define USE_ASYNC_LDS 1
typedef __attribute__((address_space(1))) v4i_t gv4i_t;   // global v4i32
typedef __attribute__((address_space(3))) v4i_t lv4i_t;   // LDS v4i32
#else
#define USE_ASYNC_LDS 0
#endif

// Monotonic float<->uint keys so we can use integer atomicMax for f32 max.
__device__ __forceinline__ unsigned ordKey(float f) {
  unsigned u = __float_as_uint(f);
  return (u & 0x80000000u) ? ~u : (u | 0x80000000u);
}
__device__ __forceinline__ float unordKey(unsigned k) {
  unsigned u = (k & 0x80000000u) ? (k & 0x7FFFFFFFu) : ~k;
  return __uint_as_float(u);
}

// ---------------------------------------------------------------------------
__global__ void init_ws_kernel(float* __restrict__ asum, unsigned* __restrict__ mx, int n) {
  int i = blockIdx.x * blockDim.x + threadIdx.x;
  if (i < 8) mx[i] = 0u;          // 0 < ordKey(any float) -> acts as -inf
  if (i < n) asum[i] = 0.0f;
}

// ---------------------------------------------------------------------------
// Projection: proj[m][n] = sum_k x[m][k] * W[n][k],  m in [0, B*N), n in [0,128)
// One wave computes one 16x16 tile. block=(32,4): 4 waves -> 4 n-tiles.
// grid = (M/16, 2). Also seeds out = proj (residual term).
__global__ __launch_bounds__(128)
void proj_wmma_kernel(const float* __restrict__ x, const float* __restrict__ W,
                      float* __restrict__ proj, float* __restrict__ out) {
  __shared__ __align__(16) float ldsW[4][16 * DD];   // 32 KB: per-wave W slice
  const int m0   = blockIdx.x * 16;
  const int nt   = threadIdx.y;
  const int n0   = (blockIdx.y * 4 + nt) * 16;
  const int lane = threadIdx.x;        // 0..31, wave32
  const int half = lane >> 4;          // 0 or 1
  const int lrow = lane & 15;

  // Stage this wave's 16x128 fp32 slice of W (8 KB) into LDS.
  // Each lane moves 16 bytes per issue, 16 issues -> 8 KB per wave.
  {
    const char* gsrc = (const char*)(W + (size_t)n0 * DD) + lane * 16;
    char*       ldst = (char*)(&ldsW[nt][0]) + lane * 16;
#if USE_ASYNC_LDS
#pragma unroll
    for (int i = 0; i < 16; ++i) {
      __builtin_amdgcn_global_load_async_to_lds_b128(
          (gv4i_t*)(gsrc + i * 512), (lv4i_t*)(ldst + i * 512), 0, 0);
    }
    __builtin_amdgcn_s_wait_asynccnt(0);   // self-consume: no cross-wave barrier
#else
#pragma unroll
    for (int i = 0; i < 16; ++i)
      *(float4*)(ldst + i * 512) = *(const float4*)(gsrc + i * 512);
#endif
  }

  v8f acc = {};
  const float* arow_base = x + (size_t)(m0 + lrow) * DD;   // A row (M = lrow)
  const float* wrow_lds  = &ldsW[nt][lrow * DD];           // B col (N = lrow)

#pragma unroll
  for (int k0 = 0; k0 < DD; k0 += 32) {
    // A fragment, 16x32 bf16: lane half 0 holds K = k0+[0..15];
    // half 1 holds K = k0+[16..31].
    v16bf a, b;
    const float* ap = arow_base + k0 + half * 16;
#pragma unroll
    for (int i = 0; i < 8; ++i) {
      a[i]     = (__bf16)ap[i];
      a[i + 8] = (__bf16)ap[i + 8];
    }
    // B fragment, 32x16 bf16: element i <-> K = k0 + 16*half + i, column = lrow.
    const float* wp = wrow_lds + k0 + half * 16;
#pragma unroll
    for (int i = 0; i < 16; ++i) b[i] = (__bf16)wp[i];

    acc = __builtin_amdgcn_wmma_f32_16x16x32_bf16(
        /*neg_a=*/false, a, /*neg_b=*/false, b,
        /*c_mod=*/(short)0, acc, /*reuse_a=*/false, /*reuse_b=*/false);
  }

  // D layout: VGPR r holds (M = r + 8*half, N = lrow).
#pragma unroll
  for (int r = 0; r < 8; ++r) {
    size_t idx = (size_t)(m0 + r + half * 8) * DD + (n0 + lrow);
    float v = acc[r];
    proj[idx] = v;
    out[idx]  = v;   // residual seed; edge messages atomically accumulate on top
  }
}

// ---------------------------------------------------------------------------
// Per-(b,n,h) attention scalars: as = <h, attn_src[h]>, ad = <h, attn_dst[h]>.
__global__ void node_alpha_kernel(const float* __restrict__ proj,
                                  const float* __restrict__ attn_src,
                                  const float* __restrict__ attn_dst,
                                  float* __restrict__ as, float* __restrict__ ad,
                                  int total /* = B*N*H */) {
  int i = blockIdx.x * blockDim.x + threadIdx.x;
  if (i >= total) return;
  int h    = i & (HH - 1);
  int node = i >> 2;                 // b*N + n
  const float* hp = proj + (size_t)node * DD + h * HDD;
  const float* sv = attn_src + h * HDD;
  const float* dv = attn_dst + h * HDD;
  float s1 = 0.f, s2 = 0.f;
#pragma unroll
  for (int d = 0; d < HDD; ++d) {
    float v = hp[d];
    s1 = fmaf(v, sv[d], s1);
    s2 = fmaf(v, dv[d], s2);
  }
  as[i] = s1;
  ad[i] = s2;
}

// ---------------------------------------------------------------------------
// Global max of leakyrelu(as[src]+ad[dst]) per (b,h): 8 slots.
__global__ __launch_bounds__(256)
void edge_max_kernel(const int* __restrict__ ei,
                     const float* __restrict__ as, const float* __restrict__ ad,
                     unsigned* __restrict__ mx, int E, int N) {
  float m[8];
#pragma unroll
  for (int s = 0; s < 8; ++s) m[s] = -INFINITY;

  int stride = gridDim.x * blockDim.x;
  for (int e = blockIdx.x * blockDim.x + threadIdx.x; e < E; e += stride) {
    // software prefetch of next iteration's indices (global_prefetch_b8)
    __builtin_prefetch(ei + e + stride, 0, 0);
    __builtin_prefetch(ei + E + e + stride, 0, 0);
    int sN = ei[e];
    int dN = ei[E + e];
#pragma unroll
    for (int b = 0; b < 2; ++b) {
      const float* sp = as + ((size_t)b * N + sN) * HH;
      const float* dp = ad + ((size_t)b * N + dN) * HH;
#pragma unroll
      for (int h = 0; h < HH; ++h) {
        float a = sp[h] + dp[h];
        a = (a >= 0.f) ? a : 0.2f * a;
        m[b * HH + h] = fmaxf(m[b * HH + h], a);
      }
    }
  }
  // wave32 shuffle reduce per slot
#pragma unroll
  for (int s = 0; s < 8; ++s)
#pragma unroll
    for (int o = 16; o > 0; o >>= 1) m[s] = fmaxf(m[s], __shfl_xor(m[s], o, 32));

  __shared__ float red[8][8];        // [wave][slot], 256 threads -> 8 waves
  int w = threadIdx.x >> 5, lane = threadIdx.x & 31;
  if (lane == 0)
#pragma unroll
    for (int s = 0; s < 8; ++s) red[w][s] = m[s];
  __syncthreads();
  if (threadIdx.x < 8) {
    float v = -INFINITY;
#pragma unroll
    for (int ww = 0; ww < 8; ++ww) v = fmaxf(v, red[ww][threadIdx.x]);
    atomicMax(&mx[threadIdx.x], ordKey(v));
  }
}

// ---------------------------------------------------------------------------
// Softmax denominator: asum[b,dst,h] += exp(lrelu(as+ad) - max[b,h])
__global__ void alpha_sum_kernel(const int* __restrict__ ei,
                                 const float* __restrict__ as, const float* __restrict__ ad,
                                 const unsigned* __restrict__ mx,
                                 float* __restrict__ asum, int E, int N) {
  int i = blockIdx.x * blockDim.x + threadIdx.x;   // over B*E*H
  if (i >= 2 * E * HH) return;
  int h = i & (HH - 1);
  int r = i >> 2;
  int b = r / E;
  int e = r - b * E;
  int sN = ei[e];
  int dN = ei[E + e];
  float a = as[((size_t)b * N + sN) * HH + h] + ad[((size_t)b * N + dN) * HH + h];
  a = (a >= 0.f) ? a : 0.2f * a;
  float ex = __expf(a - unordKey(mx[b * HH + h]));
  unsafeAtomicAdd(&asum[((size_t)b * N + dN) * HH + h], ex);
}

// ---------------------------------------------------------------------------
// Message scatter: one wave per (b, edge); lane c*32+l handles dim, head = c.
__global__ __launch_bounds__(256)
void scatter_msg_kernel(const int* __restrict__ ei, const float* __restrict__ proj,
                        const float* __restrict__ as, const float* __restrict__ ad,
                        const float* __restrict__ asum, const unsigned* __restrict__ mx,
                        float* __restrict__ out, int E, int N) {
  int gid  = blockIdx.x * blockDim.x + threadIdx.x;
  int w    = gid >> 5;
  int lane = gid & 31;
  if (w >= 2 * E) return;
  int b = w / E;
  int e = w - b * E;
  int sN = ei[e];
  int dN = ei[E + e];

  const float* sp   = as   + ((size_t)b * N + sN) * HH;
  const float* dp   = ad   + ((size_t)b * N + dN) * HH;
  const float* sump = asum + ((size_t)b * N + dN) * HH;
  float alpha[HH];
#pragma unroll
  for (int h = 0; h < HH; ++h) {
    float a = sp[h] + dp[h];
    a = (a >= 0.f) ? a : 0.2f * a;
    alpha[h] = __expf(a - unordKey(mx[b * HH + h])) / (sump[h] + 1e-8f);
  }
  const float* hsrc = proj + ((size_t)b * N + sN) * DD;
  float*       orow = out  + ((size_t)b * N + dN) * DD;
#pragma unroll
  for (int c = 0; c < 4; ++c) {
    int dim = lane + 32 * c;                 // head = c (HD == 32 == wave width)
    unsafeAtomicAdd(&orow[dim], hsrc[dim] * alpha[c]);
  }
}

// ---------------------------------------------------------------------------
// In-place LayerNorm(128) + ReLU. One wave per row; lane holds a float4.
__global__ __launch_bounds__(256)
void ln_relu_kernel(float* __restrict__ out, const float* __restrict__ gamma,
                    const float* __restrict__ beta, int rows) {
  int gid  = blockIdx.x * blockDim.x + threadIdx.x;
  int row  = gid >> 5;
  int lane = gid & 31;
  if (row >= rows) return;
  float* rp = out + (size_t)row * DD;
  float4 v = ((const float4*)rp)[lane];
  float s  = v.x + v.y + v.z + v.w;
  float ss = v.x * v.x + v.y * v.y + v.z * v.z + v.w * v.w;
#pragma unroll
  for (int o = 16; o > 0; o >>= 1) {
    s  += __shfl_xor(s, o, 32);
    ss += __shfl_xor(ss, o, 32);
  }
  float mean = s * (1.0f / DD);
  float var  = ss * (1.0f / DD) - mean * mean;
  float inv  = rsqrtf(var + 1e-5f);
  float4 g  = ((const float4*)gamma)[lane];
  float4 bt = ((const float4*)beta)[lane];
  float4 r;
  r.x = fmaxf((v.x - mean) * inv * g.x + bt.x, 0.f);
  r.y = fmaxf((v.y - mean) * inv * g.y + bt.y, 0.f);
  r.z = fmaxf((v.z - mean) * inv * g.z + bt.z, 0.f);
  r.w = fmaxf((v.w - mean) * inv * g.w + bt.w, 0.f);
  ((float4*)rp)[lane] = r;
}

// ---------------------------------------------------------------------------
extern "C" void kernel_launch(void* const* d_in, const int* in_sizes, int n_in,
                              void* d_out, int out_size, void* d_ws, size_t ws_size,
                              hipStream_t stream) {
  const float* x        = (const float*)d_in[0];
  const int*   ei       = (const int*)  d_in[1];   // jax default int32 (x64 off)
  const float* W        = (const float*)d_in[2];
  const float* attn_src = (const float*)d_in[3];
  const float* attn_dst = (const float*)d_in[4];
  const float* gamma    = (const float*)d_in[5];
  const float* beta     = (const float*)d_in[6];
  float* out = (float*)d_out;

  const int BN = in_sizes[0] / DD;   // B*N = 20000
  const int E  = in_sizes[1] / 2;    // 640000
  const int N  = BN / 2;             // B = 2

  // Workspace layout (floats): proj | as | ad | asum | mx(8 uints)
  float*    proj = (float*)d_ws;
  float*    as   = proj + (size_t)BN * DD;
  float*    ad   = as   + (size_t)BN * HH;
  float*    asum = ad   + (size_t)BN * HH;
  unsigned* mx   = (unsigned*)(asum + (size_t)BN * HH);

  // 1. init denominators + max slots
  init_ws_kernel<<<(BN * HH + 255) / 256, 256, 0, stream>>>(asum, mx, BN * HH);

  // 2. WMMA projection (also seeds out = proj for the residual)
  dim3 gA(BN / 16, 2), bA(32, 4);
  proj_wmma_kernel<<<gA, bA, 0, stream>>>(x, W, proj, out);

  // 3. per-node attention scalars
  node_alpha_kernel<<<(BN * HH + 255) / 256, 256, 0, stream>>>(
      proj, attn_src, attn_dst, as, ad, BN * HH);

  // 4. global per-(b,h) max over edges
  edge_max_kernel<<<1024, 256, 0, stream>>>(ei, as, ad, mx, E, N);

  // 5. softmax denominators (scatter)
  int totC = 2 * E * HH;
  alpha_sum_kernel<<<(totC + 255) / 256, 256, 0, stream>>>(ei, as, ad, mx, asum, E, N);

  // 6. message scatter-add: one wave per (b, edge)
  long long thr = 2LL * E * 32;
  scatter_msg_kernel<<<(int)((thr + 255) / 256), 256, 0, stream>>>(
      ei, proj, as, ad, asum, mx, out, E, N);

  // 7. LayerNorm + ReLU in place
  ln_relu_kernel<<<(BN * 32 + 255) / 256, 256, 0, stream>>>(out, gamma, beta, BN);
}
